// PartialLikelihood_64639257805423
// MI455X (gfx1250) — matrix operand order repaired and checked by reading, not compiled, not run on hardware
//
#include <hip/hip_runtime.h>
#include <hip/hip_bf16.h>

#define NN 8192
#define EPSF 1e-8f

typedef __attribute__((ext_vector_type(16))) __bf16    v16bf;
typedef __attribute__((ext_vector_type(8)))  float     v8f;
typedef __attribute__((ext_vector_type(8)))  unsigned  v8u;

__device__ __forceinline__ unsigned bf16_rne(float x) {
    unsigned u = __float_as_uint(x);
    return (u + 0x7FFFu + ((u >> 16) & 1u)) >> 16;
}

// ---------------- Kernel 1: max(risk), sum(e) ----------------
__global__ __launch_bounds__(1024) void k_reduce_max_esum(const float* __restrict__ r,
                                                          const float* __restrict__ y,
                                                          float* __restrict__ out2) {
    __shared__ float smax[1024];
    __shared__ float ssum[1024];
    int tid = threadIdx.x;
    float m = -3.402823466e+38f, s = 0.f;
    for (int i = tid; i < NN; i += 1024) {
        m = fmaxf(m, r[i]);
        s += y[2 * i + 1];
    }
    smax[tid] = m; ssum[tid] = s;
    __syncthreads();
    for (int st = 512; st > 0; st >>= 1) {
        if (tid < st) {
            smax[tid] = fmaxf(smax[tid], smax[tid + st]);
            ssum[tid] += ssum[tid + st];
        }
        __syncthreads();
    }
    if (tid == 0) { out2[0] = smax[0]; out2[1] = ssum[0]; }
}

// ---------------- Kernel 2: packed bf16 hi/lo limbs of exp(r - max) ----------------
__global__ __launch_bounds__(256) void k_prep_ex(const float* __restrict__ r,
                                                 const float* __restrict__ mx_p,
                                                 unsigned* __restrict__ exh,
                                                 unsigned* __restrict__ exl) {
    int p = blockIdx.x * 256 + threadIdx.x;       // pair index
    if (p >= NN / 2) return;
    float mx = mx_p[0];
    float e0 = expf(r[2 * p] - mx);
    float e1 = expf(r[2 * p + 1] - mx);
    unsigned h0 = bf16_rne(e0), h1 = bf16_rne(e1);
    float l0 = e0 - __uint_as_float(h0 << 16);
    float l1 = e1 - __uint_as_float(h1 << 16);
    exh[p] = (h1 << 16) | h0;
    exl[p] = (bf16_rne(l1) << 16) | bf16_rne(l0);
}

// ---------------- Kernel 3: WMMA risk-set sums + per-j loss, block partials ----------------
// Wave handles 16 j's. B[k,n] = (t_i[k] >= t_j[n]) in bf16 {0,1}; A[m,k] = ex limb (rep. over m).
__global__ __launch_bounds__(128) void k_cox_wmma(const float* __restrict__ r,
                                                  const float* __restrict__ y,
                                                  const float* __restrict__ mx_p,
                                                  const unsigned* __restrict__ exh,
                                                  const unsigned* __restrict__ exl,
                                                  float* __restrict__ partials) {
    __shared__ __align__(16) unsigned char smem[65536];
    float*    s_t   = (float*)smem;                       // 8192 f32  (32 KB)
    unsigned* s_exh = (unsigned*)(smem + 32768);          // 4096 u32  (16 KB)
    unsigned* s_exl = (unsigned*)(smem + 49152);          // 4096 u32  (16 KB)

    int tid = threadIdx.x;
    for (int i = tid; i < NN; i += 128)     s_t[i]   = y[2 * i];
    for (int p = tid; p < NN / 2; p += 128) { s_exh[p] = exh[p]; s_exl[p] = exl[p]; }
    __syncthreads();

    const int wave = tid >> 5, lane = tid & 31;
    const int jbase = blockIdx.x * 64 + wave * 16;
    const int n = lane & 15;
    const float tj = s_t[jbase + n];
    const int half4 = (lane < 16) ? 0 : 4;   // A pair-index half offset
    const int kb    = (lane < 16) ? 0 : 8;   // B K base

    v8f c = {};  // fp32 accumulator (16x16), all rows identical = rss partials
    for (int ci = 0; ci < NN / 32; ++ci) {
        const int ibase = ci * 32;
        const int pbase = ibase >> 1;
        v8u ah, al, bm;
#pragma unroll
        for (int v = 0; v < 8; ++v) {
            // A (16-bit 16x32): lane<16 K={2v,2v+1}(+16 for v>=4); lane>=16 +8
            int poff = pbase + v + ((v >= 4) ? 4 : 0) + half4;
            ah[v] = s_exh[poff];
            al[v] = s_exl[poff];
            // B (32x16): lane<16 K={2v,2v+1}(+16 for v>=4); lane>=16 +8; n = lane&15
            int klo = kb + 2 * v + ((v >= 4) ? 8 : 0);
            float t0 = s_t[ibase + klo];
            float t1 = s_t[ibase + klo + 1];
            unsigned m = 0u;
            if (t0 >= tj) m  = 0x00003F80u;   // bf16 1.0 in low half
            if (t1 >= tj) m |= 0x3F800000u;   // bf16 1.0 in high half
            bm[v] = m;
        }
        v16bf a_hi = __builtin_bit_cast(v16bf, ah);
        v16bf a_lo = __builtin_bit_cast(v16bf, al);
        v16bf b    = __builtin_bit_cast(v16bf, bm);
        c = __builtin_amdgcn_wmma_f32_16x16x32_bf16(false, a_hi, false, b, (short)0, c, false, false);
        c = __builtin_amdgcn_wmma_f32_16x16x32_bf16(false, a_lo, false, b, (short)0, c, false, false);
    }

    // C/D layout: element 0 = row M=0 (lanes 0-15) / M=8 (lanes 16-31); rows identical.
    float rss = c[0];
    float contrib = 0.f;
    if (lane < 16) {
        int j = jbase + n;
        float log_loss = logf(rss + EPSF) + mx_p[0];
        contrib = (r[j] - log_loss) * y[2 * j + 1];
    }
    __syncthreads();
    float* s_red = (float*)smem;   // reuse LDS for block reduction
    s_red[tid] = contrib;
    __syncthreads();
    for (int st = 64; st > 0; st >>= 1) {
        if (tid < st) s_red[tid] += s_red[tid + st];
        __syncthreads();
    }
    if (tid == 0) partials[blockIdx.x] = s_red[0];
}

// ---------------- Kernel 4: final scalar ----------------
__global__ __launch_bounds__(128) void k_final(const float* __restrict__ partials,
                                               const float* __restrict__ mx_esum,
                                               float* __restrict__ out) {
    __shared__ float s[128];
    int tid = threadIdx.x;
    s[tid] = partials[tid];
    __syncthreads();
    for (int st = 64; st > 0; st >>= 1) {
        if (tid < st) s[tid] += s[tid + st];
        __syncthreads();
    }
    if (tid == 0) out[0] = -(s[0] / mx_esum[1]) / (float)NN;
}

extern "C" void kernel_launch(void* const* d_in, const int* in_sizes, int n_in,
                              void* d_out, int out_size, void* d_ws, size_t ws_size,
                              hipStream_t stream) {
    const float* r = (const float*)d_in[0];   // risk_pred (8192)
    const float* y = (const float*)d_in[1];   // y_true (8192 x 2): [t, e]
    float* wsf = (float*)d_ws;
    float* partials = wsf + 16;                                   // 128 floats @ +64B
    unsigned* exh = (unsigned*)((char*)d_ws + 1024);              // 16 KB
    unsigned* exl = (unsigned*)((char*)d_ws + 1024 + 16384);      // 16 KB

    k_reduce_max_esum<<<1, 1024, 0, stream>>>(r, y, wsf);
    k_prep_ex<<<(NN / 2 + 255) / 256, 256, 0, stream>>>(r, wsf, exh, exl);
    k_cox_wmma<<<NN / 64, 128, 0, stream>>>(r, y, wsf, exh, exl, partials);
    k_final<<<1, 128, 0, stream>>>(partials, wsf, (float*)d_out);
}
